// GATLayer_84954453115320
// MI455X (gfx1250) — matrix operand order repaired and checked by reading, not compiled, not run on hardware
//
#include <hip/hip_runtime.h>

// GAT layer for MI455X (gfx1250, wave32, WMMA, async LDS copies).
// Sizes fixed by the reference: B=16, N=1024, IN_DIM=128, H=4, D=64.
#define NN   1024
#define BB   16
#define IND  128
#define HH   4
#define DD   64
#define HD   (HH*DD)          // 256
#define NEG_SLOPE 0.2f
#define NEG_BIG  (-1.0e30f)
#define HS_STRIDE 40          // LDS tile row stride in bf16 (pad 32->40, kills 2^n bank pattern)

typedef __attribute__((ext_vector_type(16))) __bf16 v16bf;
typedef __attribute__((ext_vector_type(8)))  __bf16 v8bf;
typedef __attribute__((ext_vector_type(8)))  float  v8f;

// Native bf16 conversion: gfx1250 has hardware bf16 cvt; let clang pick it.
__device__ __forceinline__ __bf16 f2bf(float f) { return (__bf16)f; }

// ---------------------------------------------------------------------------
// Prep 1: W (128x256 f32, row-major k,col) -> Wt (256x128 bf16, col-major)
// ---------------------------------------------------------------------------
__global__ __launch_bounds__(256) void pack_w_kernel(const float* __restrict__ W,
                                                     __bf16* __restrict__ Wt) {
  int idx = blockIdx.x * 256 + threadIdx.x;       // 32768 = 256*128
  int col = idx >> 7, k = idx & 127;
  Wt[col * IND + k] = f2bf(W[k * HD + col]);
}

// ---------------------------------------------------------------------------
// Prep 2: adj (N x N int32) -> bitmask, 32 edges per u32 word.
// ---------------------------------------------------------------------------
__global__ __launch_bounds__(256) void pack_adj_kernel(const int* __restrict__ adj,
                                                       unsigned* __restrict__ bits) {
  int idx = blockIdx.x * 256 + threadIdx.x;       // 32768 = 1024*32
  int i = idx >> 5, w = idx & 31;
  const int* p = adj + i * NN + w * 32;
  unsigned m = 0u;
#pragma unroll
  for (int s = 0; s < 32; ++s) m |= (p[s] != 0 ? 1u : 0u) << s;
  bits[idx] = m;
}

// ---------------------------------------------------------------------------
// Kernel A: h = x @ W via v_wmma_f32_16x16x32_bf16, fused with
//   h_t[b][head][d][n] (bf16, n contiguous) and e_src/e_dst dot products.
// Block = 128 threads = 4 waves; wave w owns head w; one 16-row tile / block.
// ---------------------------------------------------------------------------
__global__ __launch_bounds__(128) void gemm_h_kernel(
    const float* __restrict__ x, const __bf16* __restrict__ Wt,
    const float* __restrict__ a_src, const float* __restrict__ a_dst,
    __bf16* __restrict__ h_t, float* __restrict__ e_src, float* __restrict__ e_dst)
{
  const int head = threadIdx.x >> 5;
  const int lane = threadIdx.x & 31;
  const int half = lane >> 4;
  const int lm   = lane & 15;
  const int b    = blockIdx.x >> 6;            // 64 row-tiles per batch
  const int nb   = (blockIdx.x & 63) * 16;     // n base within batch
  const long rowTile = (long)blockIdx.x * 16;  // flat (b*N + n) row base

  const float* xrow = x + (rowTile + lm) * IND;

  v8f acc[4] = {};
#pragma unroll
  for (int kc = 0; kc < 4; ++kc) {
    // A tile 16x32 bf16: half0 slots -> K {0..7,16..23}, half1 -> K {8..15,24..31}
    const float* a0 = xrow + kc * 32 + half * 8;
    const float* a1 = xrow + kc * 32 + 16 + half * 8;
    float4 r0 = *(const float4*)(a0);
    float4 r1 = *(const float4*)(a0 + 4);
    float4 r2 = *(const float4*)(a1);
    float4 r3 = *(const float4*)(a1 + 4);
    v16bf amat;
    amat[0]=f2bf(r0.x); amat[1]=f2bf(r0.y); amat[2]=f2bf(r0.z); amat[3]=f2bf(r0.w);
    amat[4]=f2bf(r1.x); amat[5]=f2bf(r1.y); amat[6]=f2bf(r1.z); amat[7]=f2bf(r1.w);
    amat[8]=f2bf(r2.x); amat[9]=f2bf(r2.y); amat[10]=f2bf(r2.z); amat[11]=f2bf(r2.w);
    amat[12]=f2bf(r3.x); amat[13]=f2bf(r3.y); amat[14]=f2bf(r3.z); amat[15]=f2bf(r3.w);
#pragma unroll
    for (int t = 0; t < 4; ++t) {
      // B tile 32x16 bf16: lane = col, slots 0..7 -> K = half*8 + s, 8..15 -> K = 16 + half*8 + (s-8)
      const __bf16* bp = Wt + (head * 64 + t * 16 + lm) * IND + kc * 32 + half * 8;
      v8bf blo = *(const v8bf*)bp;
      v8bf bhi = *(const v8bf*)(bp + 16);
      v16bf bmat = __builtin_shufflevector(blo, bhi, 0,1,2,3,4,5,6,7,8,9,10,11,12,13,14,15);
      acc[t] = __builtin_amdgcn_wmma_f32_16x16x32_bf16(
          false, amat, false, bmat, (short)0, acc[t], false, false);
    }
  }

  // D layout: lane holds rows M = v + 8*half (v=0..7, contiguous n), col d = t*16+lm
  float ps[8] = {}, pd[8] = {};
#pragma unroll
  for (int t = 0; t < 4; ++t) {
    const int d = t * 16 + lm;
    const float as = a_src[head * 64 + d];
    const float ad = a_dst[head * 64 + d];
    v8bf pack;
#pragma unroll
    for (int v = 0; v < 8; ++v) {
      float hval = acc[t][v];
      ps[v] += hval * as;
      pd[v] += hval * ad;
      pack[v] = f2bf(hval);
    }
    __bf16* hp = h_t + (long)((b * HH + head) * DD + d) * NN + nb + half * 8;
    *(v8bf*)hp = pack;                        // 8 contiguous bf16 = 16B store
  }
  // reduce e-partials across the 16-lane half (cols of this head)
#pragma unroll
  for (int off = 1; off < 16; off <<= 1) {
#pragma unroll
    for (int v = 0; v < 8; ++v) {
      ps[v] += __shfl_xor(ps[v], off, 32);
      pd[v] += __shfl_xor(pd[v], off, 32);
    }
  }
  if (lm == 0) {
    float* es = e_src + (b * HH + head) * NN + nb + half * 8;
    float* ed = e_dst + (b * HH + head) * NN + nb + half * 8;
#pragma unroll
    for (int v = 0; v < 8; ++v) { es[v] = ps[v]; ed[v] = pd[v]; }
  }
}

// ---------------------------------------------------------------------------
// Kernel B: flash-style masked softmax(alpha) @ h, one wave per (b,head,16 rows).
// Block = 256 threads (8 waves). The 64x32 bf16 h-tile (4KB) is staged into LDS
// once per block with gfx1250 async copies (global_load_async_to_lds_b128,
// ASYNCcnt), double-buffered against the WMMA consumption from LDS.
// ---------------------------------------------------------------------------
__global__ __launch_bounds__(256) void gat_attn_kernel(
    const __bf16* __restrict__ h_t, const float* __restrict__ e_src_g,
    const float* __restrict__ e_dst_g, const unsigned* __restrict__ adjbits,
    float* __restrict__ out)
{
  __shared__ float  eds[NN];                    // e_dst row for this (b,head), 4KB
  __shared__ __bf16 hs[2][DD * HS_STRIDE];      // double-buffered h tile, 2 x 5KB

  const int b    = blockIdx.x >> 5;
  const int head = (blockIdx.x >> 3) & 3;
  const int ib   = blockIdx.x & 7;
  const int tid  = threadIdx.x;
  const int wave = tid >> 5;
  const int lane = tid & 31;
  const int half = lane >> 4;
  const int lm   = lane & 15;

  const __bf16* hbase = h_t + (long)((b * HH + head) * DD) * NN;

  // stage e_dst row: 1024 f32 = 256 float4, one per thread
  ((float4*)eds)[tid] = ((const float4*)(e_dst_g + (b * HH + head) * NN))[tid];

  // async-stage first h tile: thread -> (d = tid/4, 8-col chunk c = (tid%4)*8), 16B each
  const int sd = tid >> 2;
  const int sc = (tid & 3) * 8;
  {
    unsigned ldsoff = (unsigned)(uintptr_t)(&hs[0][sd * HS_STRIDE + sc]);
    unsigned long long gaddr =
        (unsigned long long)(uintptr_t)(hbase + (long)sd * NN + sc);
    asm volatile("global_load_async_to_lds_b128 %0, %1, off"
                 :: "v"(ldsoff), "v"(gaddr) : "memory");
  }
  asm volatile("s_wait_asynccnt 0x0" ::: "memory");
  __syncthreads();

  const int i0   = ib * 128 + wave * 16;     // first attention row of this wave
  const int irow = i0 + lm;                  // this lane's row i
  const float es_i = e_src_g[(b * HH + head) * NN + irow];
  const unsigned* abRow = adjbits + irow * 32;

  v8f acc[4] = {};
  float m = NEG_BIG, l = 0.0f;
  int cur = 0;

  for (int j0 = 0; j0 < NN; j0 += 32) {
    // kick off async copy of the next tile into the other buffer
    if (j0 + 32 < NN) {
      unsigned ldsoff = (unsigned)(uintptr_t)(&hs[cur ^ 1][sd * HS_STRIDE + sc]);
      unsigned long long gaddr =
          (unsigned long long)(uintptr_t)(hbase + (long)sd * NN + (j0 + 32) + sc);
      asm volatile("global_load_async_to_lds_b128 %0, %1, off"
                   :: "v"(ldsoff), "v"(gaddr) : "memory");
    }

    const unsigned w = abRow[j0 >> 5];
    // e_dst in A-layout slot order: slots 0..7 -> j0+half*8.., 8..15 -> j0+16+half*8..
    float4 e0 = *(const float4*)&eds[j0 + half * 8];
    float4 e1 = *(const float4*)&eds[j0 + half * 8 + 4];
    float4 e2 = *(const float4*)&eds[j0 + 16 + half * 8];
    float4 e3 = *(const float4*)&eds[j0 + 16 + half * 8 + 4];
    float ev[16] = { e0.x,e0.y,e0.z,e0.w, e1.x,e1.y,e1.z,e1.w,
                     e2.x,e2.y,e2.z,e2.w, e3.x,e3.y,e3.z,e3.w };
    float pe[16]; unsigned mb[16];
    float tm = NEG_BIG;
#pragma unroll
    for (int s = 0; s < 16; ++s) {
      const int K = (s < 8) ? (half * 8 + s) : (16 + half * 8 + (s - 8));
      const unsigned bit = (w >> K) & 1u;
      float e = es_i + ev[s];
      e = fmaxf(e, NEG_SLOPE * e);           // leaky relu (0<slope<1)
      pe[s] = e; mb[s] = bit;
      tm = fmaxf(tm, bit ? e : NEG_BIG);
    }
    tm = fmaxf(tm, __shfl_xor(tm, 16, 32));  // partner half covers other K range
    const float mnew  = fmaxf(m, tm);
    const float scale = __expf(m - mnew);
    float rs = 0.0f;
    v16bf amat;
#pragma unroll
    for (int s = 0; s < 16; ++s) {
      const float p = mb[s] ? __expf(pe[s] - mnew) : 0.0f;
      rs += p;
      amat[s] = f2bf(p);
    }
    rs += __shfl_xor(rs, 16, 32);
    l = l * scale + rs;
    // rescale accumulators: D row of element v is (v + 8*half); its combined
    // row stats live in lane (v + 8*half)
#pragma unroll
    for (int v = 0; v < 8; ++v) {
      const float sv = __shfl(scale, v + 8 * half, 32);
#pragma unroll
      for (int t = 0; t < 4; ++t) acc[t][v] *= sv;
    }
    m = mnew;
#pragma unroll
    for (int t = 0; t < 4; ++t) {
      // B tile 32x16 from LDS: lane = col d = t*16+lm, slots 0..7 -> K = half*8 + s, 8..15 -> K = 16 + half*8 + (s-8)
      const __bf16* hp = &hs[cur][(t * 16 + lm) * HS_STRIDE + half * 8];
      union { uint4 q[2]; v16bf v; } u;
      u.q[0] = *(const uint4*)hp;            // 2 x ds_load_b128
      u.q[1] = *(const uint4*)(hp + 16);
      acc[t] = __builtin_amdgcn_wmma_f32_16x16x32_bf16(
          false, amat, false, u.v, (short)0, acc[t], false, false);
    }

    // my async copy done + everyone finished reading hs[cur] before reuse
    asm volatile("s_wait_asynccnt 0x0" ::: "memory");
    __syncthreads();
    cur ^= 1;
  }

  // normalize by softmax denominator and store (B,N,H*D) f32
  float linv[8];
#pragma unroll
  for (int v = 0; v < 8; ++v) linv[v] = 1.0f / __shfl(l, v + 8 * half, 32);
#pragma unroll
  for (int t = 0; t < 4; ++t) {
    const int dcol = head * 64 + t * 16 + lm;
#pragma unroll
    for (int v = 0; v < 8; ++v) {
      const int n = i0 + half * 8 + v;
      out[(long)(b * NN + n) * HD + dcol] = acc[t][v] * linv[v];
    }
  }
}

// ---------------------------------------------------------------------------
// Host launcher. Inputs: x(f32), adj(i32), W(f32), a_src(f32), a_dst(f32).
// Workspace (~8.7 MB):
//   Wt   bf16 [256][128]     @ 0
//   h_t  bf16 [B][H][D][N]   @ 64KB
//   e_src f32 [B][H][N]      @ 64KB + 8MB
//   e_dst f32 [B][H][N]
//   adjbits u32 [N][N/32]
// ---------------------------------------------------------------------------
extern "C" void kernel_launch(void* const* d_in, const int* in_sizes, int n_in,
                              void* d_out, int out_size, void* d_ws, size_t ws_size,
                              hipStream_t stream) {
  const float* x     = (const float*)d_in[0];
  const int*   adj   = (const int*)d_in[1];
  const float* W     = (const float*)d_in[2];
  const float* a_src = (const float*)d_in[3];
  const float* a_dst = (const float*)d_in[4];
  float* out = (float*)d_out;

  char* ws = (char*)d_ws;
  __bf16*   Wt      = (__bf16*)(ws);
  __bf16*   h_t     = (__bf16*)(ws + 65536);
  float*    e_src   = (float*)(ws + 65536 + 8388608);
  float*    e_dst   = (float*)(ws + 65536 + 8388608 + 262144);
  unsigned* adjbits = (unsigned*)(ws + 65536 + 8388608 + 262144 + 262144);

  pack_w_kernel<<<128, 256, 0, stream>>>(W, Wt);
  pack_adj_kernel<<<128, 256, 0, stream>>>(adj, adjbits);
  gemm_h_kernel<<<BB * NN / 16, 128, 0, stream>>>(x, Wt, a_src, a_dst,
                                                  h_t, e_src, e_dst);
  gat_attn_kernel<<<BB * HH * (NN / 128), 256, 0, stream>>>(h_t, e_src, e_dst,
                                                            adjbits, out);
}
